// MultiHeadAttention_57037165691224
// MI455X (gfx1250) — compile-verified, matrix operand order checked by reading, not run on hardware
//
#include <hip/hip_runtime.h>
#include <hip/hip_bf16.h>

// ---------------- types ----------------
typedef __bf16 bf16_t;
typedef __attribute__((ext_vector_type(16))) __bf16        v16bf;
typedef __attribute__((ext_vector_type(8)))  float         v8f;
typedef __attribute__((ext_vector_type(4)))  unsigned int  u32x4;
typedef __attribute__((ext_vector_type(2)))  unsigned int  u32x2;
typedef __attribute__((ext_vector_type(4)))  float         f32x4;

// exact pointee type expected by the async-LDS builtins (from hipcc diagnostic)
typedef int i32x4v __attribute__((vector_size(16)));
typedef __attribute__((address_space(1))) i32x4v* as1_v4i;
typedef __attribute__((address_space(3))) i32x4v* as3_v4i;

#define WMMA_BF16(a, b, c) \
    __builtin_amdgcn_wmma_f32_16x16x32_bf16(false, (a), false, (b), (short)0, (c), false, false)

// ---------------- problem constants ----------------
constexpr int Bn   = 2;
constexpr int S    = 2048;
constexpr int D    = 1024;
constexpr int H    = 16;
constexpr int DK   = 64;
constexpr int Mtot = Bn * S;   // 4096
constexpr int Ntot = D;        // 1024
constexpr int Ktot = D;        // 1024

// GEMM tiling
constexpr int BM = 128, BN = 64, BK = 64;
constexpr int ASTR = BK + 8;   // 72 halves (144B rows, 16B-aligned)
constexpr int BSTR = BK + 8;   // 72

// Native bf16 conversion: fptrunc lowers to the hardware cvt op on gfx1250.
__device__ __forceinline__ bf16_t f2bf(float f) { return (bf16_t)f; }

union FragCast { v16bf bf; u32x4 q[2]; };
union Pack8    { u32x4 q;  bf16_t h[8]; };
union Pack4    { u32x2 d;  bf16_t h[4]; };

// ---------------- async global->LDS copy (gfx1250 ASYNCcnt path) ----------
#if __has_builtin(__builtin_amdgcn_global_load_async_to_lds_b128)
#define HAVE_ASYNC_LDS 1
__device__ __forceinline__ void async_copy_b128(const void* g, void* l) {
    __builtin_amdgcn_global_load_async_to_lds_b128((as1_v4i)g, (as3_v4i)l, 0, 0);
}
#else
#define HAVE_ASYNC_LDS 0
__device__ __forceinline__ void async_copy_b128(const void* g, void* l) {
    *(u32x4*)l = *(const u32x4*)g;
}
#endif

__device__ __forceinline__ void async_wait() {
#if HAVE_ASYNC_LDS
#if __has_builtin(__builtin_amdgcn_s_wait_asynccnt)
    __builtin_amdgcn_s_wait_asynccnt(0);
#else
    asm volatile("s_wait_asynccnt 0" ::: "memory");
#endif
#endif
}

// Fragment loader: storage row-major [row][k], STRIDE halves (multiple of 8).
// Per ISA A-layout: lane&15 = row; lanes 16-31 take K-group +8.
// Each lane's data = two contiguous 16B runs -> 2x ds_load_b128.
template <int STRIDE>
__device__ __forceinline__ v16bf load_frag(const bf16_t* base, int rowbase,
                                           int kbase, int lane) {
    int r  = rowbase + (lane & 15);
    int kb = kbase + ((lane >> 4) << 3);
    const bf16_t* p = base + r * STRIDE + kb;
    FragCast f;
    f.q[0] = *(const u32x4*)(p);        // K = kb .. kb+7
    f.q[1] = *(const u32x4*)(p + 16);   // K = kb+16 .. kb+23
    return f.bf;
}

// ======================================================================
// GEMM: out = (A[Mtot,Ktot] @ W[Ktot,Ntot] + bias) * out_scale
// A_BF16: A is bf16 (else fp32, converted to bf16 on LDS fill)
// OUT_MODE 0: fp32 plain row-major [Mtot,Ntot]
// OUT_MODE 1: bf16 split-head [B,H,S,DK]
// ======================================================================
template <bool A_BF16, int OUT_MODE>
__global__ __launch_bounds__(256) void gemm_wmma_kernel(
    const void* __restrict__ A_, const float* __restrict__ W,
    const float* __restrict__ bias, void* __restrict__ out, float out_scale) {
    __shared__ alignas(16) bf16_t As[BM][ASTR];   // [m][k]
    __shared__ alignas(16) bf16_t Bs[BN][BSTR];   // [n][k]  (W tile transposed)

    const int tid  = threadIdx.x;
    const int lane = tid & 31;
    const int w    = tid >> 5;          // 8 waves
    const int wr   = w >> 1;            // 0..3  (32-row strips)
    const int wc   = w & 1;             // 0..1  (32-col strips)
    const int m0   = blockIdx.y * BM;
    const int n0   = blockIdx.x * BN;

    const float*  Af = (const float*)A_;
    const bf16_t* Ab = (const bf16_t*)A_;

    v8f zero = {};
    v8f acc[2][2];
#pragma unroll
    for (int i = 0; i < 2; ++i)
#pragma unroll
        for (int j = 0; j < 2; ++j) acc[i][j] = zero;

    for (int k0 = 0; k0 < Ktot; k0 += BK) {
        __syncthreads();
        if (A_BF16) {
            // 128x64 bf16 = 1024 16B-chunks, 4 per thread; async DMA to LDS
#pragma unroll
            for (int i = 0; i < 4; ++i) {
                int idx = tid + 256 * i;
                int r = idx >> 3, c8 = idx & 7;
                async_copy_b128(Ab + (size_t)(m0 + r) * Ktot + k0 + c8 * 8,
                                &As[r][c8 * 8]);
            }
        } else {
            // 128x64 f32 = 2048 float4-chunks, 8 per thread (convert to bf16)
#pragma unroll
            for (int i = 0; i < 8; ++i) {
                int idx = tid + 256 * i;
                int r = idx >> 4, c4 = idx & 15;
                f32x4 v = *(const f32x4*)(Af + (size_t)(m0 + r) * Ktot + k0 + c4 * 4);
                Pack4 p;
                p.h[0] = f2bf(v.x); p.h[1] = f2bf(v.y);
                p.h[2] = f2bf(v.z); p.h[3] = f2bf(v.w);
                *(u32x2*)&As[r][c4 * 4] = p.d;
            }
        }
        // W tile 64x64 f32, coalesced read, transposed scatter into Bs[n][k]
#pragma unroll
        for (int i = 0; i < 4; ++i) {
            int idx = tid + 256 * i;
            int kr = idx >> 4, c4 = idx & 15;
            f32x4 v = *(const f32x4*)(W + (size_t)(k0 + kr) * Ntot + n0 + c4 * 4);
            Bs[c4 * 4 + 0][kr] = f2bf(v.x);
            Bs[c4 * 4 + 1][kr] = f2bf(v.y);
            Bs[c4 * 4 + 2][kr] = f2bf(v.z);
            Bs[c4 * 4 + 3][kr] = f2bf(v.w);
        }
        if (A_BF16) async_wait();
        __syncthreads();

#pragma unroll
        for (int kk = 0; kk < BK; kk += 32) {
            v16bf a0 = load_frag<ASTR>(&As[0][0], wr * 32,      kk, lane);
            v16bf a1 = load_frag<ASTR>(&As[0][0], wr * 32 + 16, kk, lane);
            v16bf b0 = load_frag<BSTR>(&Bs[0][0], wc * 32,      kk, lane);
            v16bf b1 = load_frag<BSTR>(&Bs[0][0], wc * 32 + 16, kk, lane);
            acc[0][0] = WMMA_BF16(a0, b0, acc[0][0]);
            acc[0][1] = WMMA_BF16(a0, b1, acc[0][1]);
            acc[1][0] = WMMA_BF16(a1, b0, acc[1][0]);
            acc[1][1] = WMMA_BF16(a1, b1, acc[1][1]);
        }
    }

    // epilogue
#pragma unroll
    for (int rt = 0; rt < 2; ++rt) {
#pragma unroll
        for (int ct = 0; ct < 2; ++ct) {
            int n  = n0 + wc * 32 + ct * 16 + (lane & 15);
            float bv = bias[n];
#pragma unroll
            for (int i = 0; i < 8; ++i) {
                int m = m0 + wr * 32 + rt * 16 + i + ((lane >> 4) << 3);
                float v = (acc[rt][ct][i] + bv) * out_scale;
                if (OUT_MODE == 0) {
                    ((float*)out)[(size_t)m * Ntot + n] = v;
                } else {
                    int b_ = m >> 11, s_ = m & (S - 1);
                    int h_ = n >> 6,  dk = n & (DK - 1);
                    ((bf16_t*)out)[(((size_t)(b_ * H + h_)) * S + s_) * DK + dk] =
                        f2bf(v);
                }
            }
        }
    }
}

// ======================================================================
// Flash attention: one block per (b*H+h, 64-row q tile).
// Qh/Kh/Vh: bf16 [B,H,S,DK]; Q already scaled by 1/sqrt(DK).
// Writes bf16 concat [B,S,D].
// ======================================================================
__global__ __launch_bounds__(256) void flash_attn_kernel(
    const bf16_t* __restrict__ Qh, const bf16_t* __restrict__ Kh,
    const bf16_t* __restrict__ Vh, bf16_t* __restrict__ concat,
    const int* __restrict__ causal_flag) {
    __shared__ alignas(16) bf16_t Qs[64][72];   // [q][dk]
    __shared__ alignas(16) bf16_t Ks[64][72];   // [kv][dk]  (B for QK^T: n=kv, k=dk)
    __shared__ alignas(16) bf16_t Vt[64][72];   // [dk][kv]  (B for P@V:  n=dk, k=kv)
    __shared__ alignas(16) bf16_t Ps[64][72];   // [q][kv]
    __shared__ alignas(16) float  Ssc[64][68];  // scores staging (272B rows)
    __shared__ float mrow[64], lrow[64], crow[64];

    const int tid  = threadIdx.x;
    const int lane = tid & 31;
    const int w    = tid >> 5;
    const int wr   = w >> 1;   // 0..3 -> 16-row strip
    const int wc   = w & 1;    // 0..1 -> 32-col strip

    const int qt = blockIdx.x;          // q tile index (S/64 = 32)
    const int q0 = qt * 64;
    const int bh = blockIdx.y;          // b*H + h
    const size_t head_base = (size_t)bh * S * DK;
    const bool causal = (*causal_flag != 0);

    // load Q tile (64x64 bf16 = 512 16B-chunks, 2 per thread) via async DMA
#pragma unroll
    for (int i = 0; i < 2; ++i) {
        int idx = tid + 256 * i;
        int r = idx >> 3, c8 = idx & 7;
        async_copy_b128(Qh + head_base + (size_t)(q0 + r) * DK + c8 * 8,
                        &Qs[r][c8 * 8]);
    }
    if (tid < 64) { mrow[tid] = -1e30f; lrow[tid] = 0.0f; }

    v8f zero = {};
    v8f oacc[2];
    oacc[0] = zero; oacc[1] = zero;

    const int jmax = causal ? qt : (S / 64 - 1);
    for (int j = 0; j <= jmax; ++j) {
        const int kv0 = j * 64;
        if (j == 0) async_wait();       // Q tile in LDS
        __syncthreads();
        // K tile: async DMA, natural [kv][dk]; V tile: transposed into Vt[dk][kv]
#pragma unroll
        for (int i = 0; i < 2; ++i) {
            int idx = tid + 256 * i;
            int r = idx >> 3, c8 = idx & 7;
            size_t g = head_base + (size_t)(kv0 + r) * DK + c8 * 8;
            async_copy_b128(Kh + g, &Ks[r][c8 * 8]);
            Pack8 pv;
            pv.q = *(const u32x4*)(Vh + g);
#pragma unroll
            for (int jj = 0; jj < 8; ++jj) Vt[c8 * 8 + jj][r] = pv.h[jj];
        }
        // prefetch next K/V tiles into cache while we compute
        if (j < jmax) {
            size_t gn = head_base + (size_t)(kv0 + 64) * DK;
            __builtin_prefetch(Kh + gn + (size_t)tid * 16, 0, 0);
            __builtin_prefetch(Vh + gn + (size_t)tid * 16, 0, 0);
        }
        async_wait();
        __syncthreads();

        // S = Q @ K^T  (scale folded into Q)
        {
            v8f sacc0 = zero, sacc1 = zero;
#pragma unroll
            for (int kk = 0; kk < DK; kk += 32) {
                v16bf aq = load_frag<72>(&Qs[0][0], wr * 16,      kk, lane);
                v16bf b0 = load_frag<72>(&Ks[0][0], wc * 32,      kk, lane);
                v16bf b1 = load_frag<72>(&Ks[0][0], wc * 32 + 16, kk, lane);
                sacc0 = WMMA_BF16(aq, b0, sacc0);
                sacc1 = WMMA_BF16(aq, b1, sacc1);
            }
            int n0c = wc * 32 + (lane & 15);
#pragma unroll
            for (int i = 0; i < 8; ++i) {
                int r = wr * 16 + i + ((lane >> 4) << 3);
                Ssc[r][n0c]      = sacc0[i];
                Ssc[r][n0c + 16] = sacc1[i];
            }
        }
        __syncthreads();

        // online softmax: one thread per q row, vectorized score reads
        if (tid < 64) {
            int r  = tid;
            int qi = q0 + r;
            float m_old = mrow[r];
            float tmax = -1e30f;
#pragma unroll
            for (int c4 = 0; c4 < 16; ++c4) {
                f32x4 sv = *(const f32x4*)&Ssc[r][c4 * 4];
#pragma unroll
                for (int jj = 0; jj < 4; ++jj) {
                    int c = c4 * 4 + jj;
                    bool valid = (!causal) || (kv0 + c <= qi);
                    float s = valid ? sv[jj] : -1e30f;
                    tmax = fmaxf(tmax, s);
                }
            }
            float m_new = fmaxf(m_old, tmax);
            float corr  = __expf(m_old - m_new);
            float psum  = 0.0f;
#pragma unroll
            for (int c8 = 0; c8 < 8; ++c8) {
                f32x4 s0 = *(const f32x4*)&Ssc[r][c8 * 8];
                f32x4 s1 = *(const f32x4*)&Ssc[r][c8 * 8 + 4];
                Pack8 pk;
#pragma unroll
                for (int jj = 0; jj < 8; ++jj) {
                    int c = c8 * 8 + jj;
                    bool valid = (!causal) || (kv0 + c <= qi);
                    float sv = (jj < 4) ? s0[jj] : s1[jj - 4];
                    float p  = valid ? __expf(sv - m_new) : 0.0f;
                    psum += p;
                    pk.h[jj] = f2bf(p);
                }
                *(u32x4*)&Ps[r][c8 * 8] = pk.q;
            }
            mrow[r] = m_new;
            lrow[r] = lrow[r] * corr + psum;
            crow[r] = corr;
        }
        __syncthreads();

        // rescale O accumulators, then O += P @ V
#pragma unroll
        for (int ct = 0; ct < 2; ++ct)
#pragma unroll
            for (int i = 0; i < 8; ++i) {
                int r = wr * 16 + i + ((lane >> 4) << 3);
                oacc[ct][i] *= crow[r];
            }
#pragma unroll
        for (int kk = 0; kk < 64; kk += 32) {
            v16bf ap = load_frag<72>(&Ps[0][0], wr * 16,      kk, lane);
            v16bf v0 = load_frag<72>(&Vt[0][0], wc * 32,      kk, lane);
            v16bf v1 = load_frag<72>(&Vt[0][0], wc * 32 + 16, kk, lane);
            oacc[0] = WMMA_BF16(ap, v0, oacc[0]);
            oacc[1] = WMMA_BF16(ap, v1, oacc[1]);
        }
    }
    __syncthreads();

    // write concat [B,S,D] as bf16
    const int h_ = bh & (H - 1);
    const int b_ = bh >> 4;
#pragma unroll
    for (int ct = 0; ct < 2; ++ct) {
        int dcol = h_ * DK + wc * 32 + ct * 16 + (lane & 15);
#pragma unroll
        for (int i = 0; i < 8; ++i) {
            int r = wr * 16 + i + ((lane >> 4) << 3);
            float v = oacc[ct][i] / lrow[r];
            concat[((size_t)(b_ * S) + q0 + r) * D + dcol] = f2bf(v);
        }
    }
}

// ======================================================================
// launch
// ======================================================================
extern "C" void kernel_launch(void* const* d_in, const int* in_sizes, int n_in,
                              void* d_out, int out_size, void* d_ws,
                              size_t ws_size, hipStream_t stream) {
    const float* q  = (const float*)d_in[0];
    const float* k  = (const float*)d_in[1];
    const float* v  = (const float*)d_in[2];
    const float* Wq = (const float*)d_in[3];
    const float* bq = (const float*)d_in[4];
    const float* Wk = (const float*)d_in[5];
    const float* bk = (const float*)d_in[6];
    const float* Wv = (const float*)d_in[7];
    const float* bv = (const float*)d_in[8];
    const float* Wo = (const float*)d_in[9];
    const float* bo = (const float*)d_in[10];
    const int* causal = (const int*)d_in[11];

    const size_t elems = (size_t)Bn * S * D;   // 4M elements per tensor
    bf16_t* Qh     = (bf16_t*)d_ws;
    bf16_t* Kh     = Qh + elems;
    bf16_t* Vh     = Kh + elems;
    bf16_t* concat = Vh + elems;

    dim3 ggrid(Ntot / BN, Mtot / BM);   // (16, 32)
    dim3 gblk(256);

    // projections (scale 1/sqrt(DK)=0.125 folded into Q), bf16 split-head out
    gemm_wmma_kernel<false, 1><<<ggrid, gblk, 0, stream>>>(q, Wq, bq, Qh, 0.125f);
    gemm_wmma_kernel<false, 1><<<ggrid, gblk, 0, stream>>>(k, Wk, bk, Kh, 1.0f);
    gemm_wmma_kernel<false, 1><<<ggrid, gblk, 0, stream>>>(v, Wv, bv, Vh, 1.0f);

    // attention: (q tiles, B*H)
    dim3 agrid(S / 64, Bn * H);         // (32, 32)
    flash_attn_kernel<<<agrid, gblk, 0, stream>>>(Qh, Kh, Vh, concat, causal);

    // output projection: bf16 concat -> fp32 out
    gemm_wmma_kernel<true, 0><<<ggrid, gblk, 0, stream>>>(concat, Wo, bo,
                                                          (float*)d_out, 1.0f);
}
